// RNNoiseTorch_32890859553563
// MI455X (gfx1250) — matrix operand order repaired and checked by reading, not compiled, not run on hardware
//
#include <hip/hip_runtime.h>
#include <cmath>

typedef _Float16 f16;
typedef __attribute__((ext_vector_type(16))) _Float16     v16h;
typedef __attribute__((ext_vector_type(8)))  float        v8f;
typedef __attribute__((ext_vector_type(4)))  float        v4f;
typedef __attribute__((ext_vector_type(8)))  unsigned int v8u;
typedef __attribute__((ext_vector_type(4)))  unsigned int v4u;

#define NTHREADS 256
#define TSTEPS   2000
#define BATCH    256

// ---------------- LDS layout ----------------
// fp16 region (element offsets). Weights stored padded [Npad][Kpad], row-major,
// torch [N_out][K_in] orientation so B-fragments are contiguous per lane.
// All bases/strides 16-byte aligned -> fragments load as ds_load_b128.
constexpr int W_IN   = 0;                   // in_w      24x42 -> 32x64
constexpr int W_VIH  = W_IN  + 32 * 64;     // vad_wih   72x24 -> 80x32
constexpr int W_VHH  = W_VIH + 80 * 32;     // vad_whh   72x24 -> 80x32
constexpr int W_NIH  = W_VHH + 80 * 32;     // noise_wih 144x90 -> 144x96
constexpr int W_NHH  = W_NIH + 144 * 96;    // noise_whh 144x48 -> 144x64
constexpr int W_DIH  = W_NHH + 144 * 64;    // den_wih   288x114 -> 288x128
constexpr int W_DHH  = W_DIH + 288 * 128;   // den_whh   288x96 -> 288x96
constexpr int W_OUT  = W_DHH + 288 * 96;    // out_w     22x96 -> 32x96
// fp16 activation (A-matrix) buffers, [16][Kpad] row-major, zero padded
constexpr int A_X    = W_OUT + 32 * 96;     // x_t        16x64 (42 valid)
constexpr int A_TMP  = A_X   + 16 * 64;     // tmp        16x32 (24 valid)
constexpr int A_NIN  = A_TMP + 16 * 32;     // noise_in   16x96 (90 valid: tmp|vad_h|x)
constexpr int A_DIN  = A_NIN + 16 * 96;     // den_in     16x128 (114 valid: vad_h|noise_h|x)
constexpr int A_VH   = A_DIN + 16 * 128;    // vad_h f16  16x32 (24 valid)
constexpr int A_NH   = A_VH  + 16 * 32;     // noise_h    16x64 (48 valid)
constexpr int A_DH   = A_NH  + 16 * 64;     // den_h      16x96
constexpr int HALF_COUNT = A_DH + 16 * 96;

// fp32 region (float offsets, region begins at HALF_COUNT*2 bytes).
// GEMM results and fp32 hidden state stored TRANSPOSED: buf[col*16 + m];
// linear index == (col<<4)|m, so elementwise loops use shift/mask addressing only.
constexpr int C_IN   = 0;                   // input-layer result 32x16
constexpr int C_VXP  = C_IN  + 32 * 16;     // vad xp   80x16
constexpr int C_VHP  = C_VXP + 80 * 16;     // vad hp   80x16
constexpr int C_NXP  = C_VHP + 80 * 16;     // noise xp 144x16
constexpr int C_NHP  = C_NXP + 144 * 16;    // noise hp 144x16
constexpr int C_DXP  = C_NHP + 144 * 16;    // den xp   288x16
constexpr int C_DHP  = C_DXP + 288 * 16;    // den hp   288x16
constexpr int H_V    = C_DHP + 288 * 16;    // vad_h fp32 [i*16+m]
constexpr int H_N    = H_V   + 24 * 16;     // noise_h    [i*16+m]
constexpr int H_D    = H_N   + 48 * 16;     // den_h      [i*16+m]
constexpr int B_IN   = H_D   + 96 * 16;     // biases:
constexpr int B_VIH  = B_IN  + 24;
constexpr int B_VHH  = B_VIH + 72;
constexpr int B_NIH  = B_VHH + 72;
constexpr int B_NHH  = B_NIH + 144;
constexpr int B_DIH  = B_NHH + 144;
constexpr int B_DHH  = B_DIH + 288;
constexpr int B_OUT  = B_DHH + 288;         // out_b (22)
constexpr int B_VOW  = B_OUT + 22;          // vad_out_w (24)
constexpr int B_VOB  = B_VOW + 24;          // vad_out_b (1)
constexpr int FLOAT_COUNT = B_VOB + 1;

constexpr size_t SMEM_BYTES = (size_t)HALF_COUNT * 2 + (size_t)FLOAT_COUNT * 4;
static_assert(SMEM_BYTES <= 320u * 1024u, "exceeds WGP LDS");

__device__ __forceinline__ float sigmoidf_(float x) { return 1.0f / (1.0f + __expf(-x)); }
__device__ __forceinline__ float tanhf_(float x)    { return 2.0f * sigmoidf_(2.0f * x) - 1.0f; }

// ---- WMMA fragment loaders (ISA 7.12.2 layouts, 16-bit) ----
// A 16x32: lane m = lane&15; lane-half adds K+8. Pairs 0-3 = K{0,2,4,6}(+1),
// pairs 4-7 = K{16..22}(+1): two contiguous 16B chunks -> 2x ds_load_b128.
__device__ __forceinline__ v16h fragA(const f16* base, int stride, int kbase, int lane) {
    const f16* row = base + (lane & 15) * stride + kbase + ((lane >> 4) << 3);
    v4u lo = *(const v4u*)(row);
    v4u hi = *(const v4u*)(row + 16);
    v8u u  = {lo.x, lo.y, lo.z, lo.w, hi.x, hi.y, hi.z, hi.w};
    return __builtin_bit_cast(v16h, u);
}
// B 32x16 (we hold W[N][K]; B = W^T): lane n = lane&15; lane-half adds K+8, same K order as fragA.
// two 16-byte chunks per lane (K+0..7, K+16..23) -> 2x ds_load_b128.
__device__ __forceinline__ v16h fragB(const f16* base, int stride, int kbase, int lane) {
    const f16* row = base + (lane & 15) * stride + kbase + ((lane >> 4) << 3);
    v4u lo = *(const v4u*)(row);
    v4u hi = *(const v4u*)(row + 16);
    v8u u  = {lo.x, lo.y, lo.z, lo.w, hi.x, hi.y, hi.z, hi.w};
    return __builtin_bit_cast(v16h, u);
}

// One N-tile accumulation chain over KS k-steps.
template <int KS>
__device__ __forceinline__ v8f gemm_acc(const f16* Bt, int bs, const v16h* a, int lane) {
    v8f acc = {0.f, 0.f, 0.f, 0.f, 0.f, 0.f, 0.f, 0.f};
#pragma unroll
    for (int ks = 0; ks < KS; ++ks) {
        v16h b = fragB(Bt, bs, ks * 32, lane);
        acc = __builtin_amdgcn_wmma_f32_16x16x32_f16(false, a[ks], false, b,
                                                     (short)0, acc, false, false);
    }
    return acc;
}
// Transposed D store: 8 contiguous floats per lane -> 2x ds_store_b128.
__device__ __forceinline__ void storeC(float* C, int n, int mb, v8f acc) {
    v4f lo = {acc[0], acc[1], acc[2], acc[3]};
    v4f hi = {acc[4], acc[5], acc[6], acc[7]};
    *(v4f*)(C + n * 16 + mb)     = lo;
    *(v4f*)(C + n * 16 + mb + 4) = hi;
}

// C_T[nTiles*16 x 16] = (A[16 x KS*32] * W^T)^T ; tiles split across waves.
// A-fragments hoisted (tile-invariant); two independent tiles per iteration so
// the second chain's loads/wmmas fill the first chain's WMMA->VALU hazard window.
template <int KS>
__device__ __forceinline__ void gemm_lds(const f16* A, int as, const f16* B, int bs,
                                         float* C, int nTiles, int tile0, int tileStep,
                                         int lane) {
    v16h a[KS];
#pragma unroll
    for (int ks = 0; ks < KS; ++ks) a[ks] = fragA(A, as, ks * 32, lane);
    const int nl = lane & 15;
    const int mb = (lane >> 4) << 3;
    int nt = tile0;
    for (; nt + tileStep < nTiles; nt += 2 * tileStep) {
        int nt2 = nt + tileStep;
        v8f acc0 = gemm_acc<KS>(B + nt  * 16 * bs, bs, a, lane);
        v8f acc1 = gemm_acc<KS>(B + nt2 * 16 * bs, bs, a, lane);
        storeC(C, nt  * 16 + nl, mb, acc0);
        storeC(C, nt2 * 16 + nl, mb, acc1);
    }
    if (nt < nTiles) {
        v8f acc = gemm_acc<KS>(B + nt * 16 * bs, bs, a, lane);
        storeC(C, nt * 16 + nl, mb, acc);
    }
}

__device__ __forceinline__ void loadW(const float* src, f16* dst, int N, int K,
                                      int Npad, int Kpad, int tid) {
    for (int idx = tid; idx < Npad * Kpad; idx += NTHREADS) {
        int n = idx / Kpad, k = idx - n * Kpad;
        float v = (n < N && k < K) ? src[n * K + k] : 0.0f;
        dst[idx] = (f16)v;
    }
}
__device__ __forceinline__ void copyF(const float* s, float* d, int n, int tid) {
    for (int i = tid; i < n; i += NTHREADS) d[i] = s[i];
}

// GRU elementwise on transposed buffers; each thread handles 4 consecutive m
// (one v4f chunk) -> gate reads are ds_load_b128, addressing is shift/mask only.
template <int H>
__device__ __forceinline__ void gru_update_v(const float* xp, const float* hp,
                                             const float* bih, const float* bhh,
                                             float* h32, f16* hA, int has,
                                             f16* d1, int d1s, int d1o,
                                             f16* d2, int d2s, int d2o, int tid) {
    for (int base = tid * 4; base < 16 * H; base += NTHREADS * 4) {
        const int i = base >> 4;   // column (0..H-1)
        const int m = base & 15;   // row start (0,4,8,12)
        v4f xr = *(const v4f*)(xp + base);
        v4f xz = *(const v4f*)(xp + 16 * H + base);
        v4f xn = *(const v4f*)(xp + 32 * H + base);
        v4f hr = *(const v4f*)(hp + base);
        v4f hz = *(const v4f*)(hp + 16 * H + base);
        v4f hn = *(const v4f*)(hp + 32 * H + base);
        v4f hv = *(const v4f*)(h32 + base);
        float bir = bih[i], biz = bih[H + i], bin = bih[2 * H + i];
        float bhr = bhh[i], bhz = bhh[H + i], bhn = bhh[2 * H + i];
        v4f ho;
#pragma unroll
        for (int j = 0; j < 4; ++j) {
            float r = sigmoidf_(xr[j] + bir + hr[j] + bhr);
            float z = sigmoidf_(xz[j] + biz + hz[j] + bhz);
            float n = tanhf_(xn[j] + bin + r * (hn[j] + bhn));
            ho[j] = (1.0f - z) * n + z * hv[j];
        }
        *(v4f*)(h32 + base) = ho;
#pragma unroll
        for (int j = 0; j < 4; ++j) {
            f16 hh = (f16)ho[j];
            hA[(m + j) * has + i] = hh;
            if (d1) d1[(m + j) * d1s + d1o + i] = hh;
            if (d2) d2[(m + j) * d2s + d2o + i] = hh;
        }
    }
}

__global__ void __launch_bounds__(NTHREADS) rnnoise_fused_kernel(
    const float* __restrict__ xg,
    const float* in_w, const float* in_b,
    const float* vad_wih, const float* vad_whh, const float* vad_bih, const float* vad_bhh,
    const float* vad_out_w, const float* vad_out_b,
    const float* noise_wih, const float* noise_whh, const float* noise_bih, const float* noise_bhh,
    const float* den_wih, const float* den_whh, const float* den_bih, const float* den_bhh,
    const float* out_w, const float* out_b,
    float* __restrict__ out) {
    extern __shared__ char smem[];
    f16*   Hm = (f16*)smem;
    float* Fm = (float*)(smem + (size_t)HALF_COUNT * 2);
    const int tid  = threadIdx.x;
    const int lane = tid & 31;
    const int wave = __builtin_amdgcn_readfirstlane(tid >> 5);  // provably wave-uniform
    const int b0   = blockIdx.x * 16;

    // ---- one-time: weights -> LDS fp16 (padded), biases -> LDS fp32, zero state ----
    loadW(in_w,      Hm + W_IN,   24,  42,  32,  64, tid);
    loadW(vad_wih,   Hm + W_VIH,  72,  24,  80,  32, tid);
    loadW(vad_whh,   Hm + W_VHH,  72,  24,  80,  32, tid);
    loadW(noise_wih, Hm + W_NIH, 144,  90, 144,  96, tid);
    loadW(noise_whh, Hm + W_NHH, 144,  48, 144,  64, tid);
    loadW(den_wih,   Hm + W_DIH, 288, 114, 288, 128, tid);
    loadW(den_whh,   Hm + W_DHH, 288,  96, 288,  96, tid);
    loadW(out_w,     Hm + W_OUT,  22,  96,  32,  96, tid);
    copyF(in_b,      Fm + B_IN,   24, tid);
    copyF(vad_bih,   Fm + B_VIH,  72, tid);
    copyF(vad_bhh,   Fm + B_VHH,  72, tid);
    copyF(noise_bih, Fm + B_NIH, 144, tid);
    copyF(noise_bhh, Fm + B_NHH, 144, tid);
    copyF(den_bih,   Fm + B_DIH, 288, tid);
    copyF(den_bhh,   Fm + B_DHH, 288, tid);
    copyF(out_b,     Fm + B_OUT,  22, tid);
    copyF(vad_out_w, Fm + B_VOW,  24, tid);
    copyF(vad_out_b, Fm + B_VOB,   1, tid);
    for (int i = tid; i < HALF_COUNT - A_X; i += NTHREADS) Hm[A_X + i] = (f16)0.0f; // acts+pads
    for (int i = tid; i < (24 + 48 + 96) * 16; i += NTHREADS) Fm[H_V + i] = 0.0f;   // h0 = 0
    __syncthreads();

    const size_t vadBase = (size_t)BATCH * TSTEPS * 22;

    for (int t = 0; t < TSTEPS; ++t) {
        // ---- phase 0: stream x_t, fan out to in/noise/den A-buffers (fp16) ----
        for (int idx = tid; idx < 16 * 64; idx += NTHREADS) {
            int m = idx >> 6, d = idx & 63;
            if (d < 42) {
                f16 hv = (f16)xg[((size_t)(b0 + m) * TSTEPS + t) * 42 + d];
                Hm[A_X   + m * 64  + d]      = hv;
                Hm[A_NIN + m * 96  + 48 + d] = hv;
                Hm[A_DIN + m * 128 + 72 + d] = hv;
            }
        }
        if (t + 1 < TSTEPS && tid < 16)   // global_prefetch_b8 for next step's rows
            __builtin_prefetch(&xg[((size_t)(b0 + tid) * TSTEPS + (t + 1)) * 42], 0, 1);
        __syncthreads();

        // ---- phase 1: input dense (waves 0,1) || vad hidden (waves 2..6),
        //      plus den hidden proj on h_{t-1} (all waves, extra tiles on waves
        //      6,7; wave 7 carries no other work here) — off the dependency chain.
        if (wave < 2)
            gemm_lds<2>(Hm + A_X,  64, Hm + W_IN,  64, Fm + C_IN,  2, wave,     2, lane);
        else
            gemm_lds<1>(Hm + A_VH, 32, Hm + W_VHH, 32, Fm + C_VHP, 5, wave - 2, 6, lane);
        gemm_lds<3>(Hm + A_DH, 96, Hm + W_DHH, 96, Fm + C_DHP, 18, 7 - wave, 8, lane);
        __syncthreads();

        // ---- phase 2: tmp = tanh(.) (vectorized, shift/mask addressing) ----
        for (int base = tid * 4; base < 16 * 24; base += NTHREADS * 4) {
            int i = base >> 4, m = base & 15;
            v4f c = *(const v4f*)(Fm + C_IN + base);
            float bi = Fm[B_IN + i];
#pragma unroll
            for (int j = 0; j < 4; ++j) {
                f16 hv = (f16)tanhf_(c[j] + bi);
                Hm[A_TMP + (m + j) * 32 + i] = hv;
                Hm[A_NIN + (m + j) * 96 + i] = hv;
            }
        }
        __syncthreads();

        // ---- phase 3: vad input proj (waves 0-4) + noise hidden proj on
        //      h_{t-1} (extra tile on wave 7) ----
        gemm_lds<1>(Hm + A_TMP, 32, Hm + W_VIH, 32, Fm + C_VXP, 5, wave,     8, lane);
        gemm_lds<2>(Hm + A_NH,  64, Hm + W_NHH, 64, Fm + C_NHP, 9, 7 - wave, 8, lane);
        __syncthreads();

        // ---- phase 4: vad GRU update (feeds noise_in[24:48], den_in[0:24]) ----
        gru_update_v<24>(Fm + C_VXP, Fm + C_VHP, Fm + B_VIH, Fm + B_VHH,
                         Fm + H_V, Hm + A_VH, 32,
                         Hm + A_NIN, 96, 24, Hm + A_DIN, 128, 0, tid);
        __syncthreads();

        // ---- phase 5: noise input proj ----
        gemm_lds<3>(Hm + A_NIN, 96, Hm + W_NIH, 96, Fm + C_NXP, 9, wave, 8, lane);
        __syncthreads();

        // ---- phase 6: noise GRU update (feeds den_in[24:72]) + vad head ----
        gru_update_v<48>(Fm + C_NXP, Fm + C_NHP, Fm + B_NIH, Fm + B_NHH,
                         Fm + H_N, Hm + A_NH, 64,
                         Hm + A_DIN, 128, 24, (f16*)nullptr, 0, 0, tid);
        if (tid < 16) {
            float acc = Fm[B_VOB];
            for (int i = 0; i < 24; ++i) acc += Fm[H_V + i * 16 + tid] * Fm[B_VOW + i];
            out[vadBase + (size_t)(b0 + tid) * TSTEPS + t] = sigmoidf_(acc);
        }
        __syncthreads();

        // ---- phase 7: den input proj (18 tiles over 8 waves) ----
        gemm_lds<4>(Hm + A_DIN, 128, Hm + W_DIH, 128, Fm + C_DXP, 18, wave, 8, lane);
        __syncthreads();

        // ---- phase 8: den GRU update ----
        gru_update_v<96>(Fm + C_DXP, Fm + C_DHP, Fm + B_DIH, Fm + B_DHH,
                         Fm + H_D, Hm + A_DH, 96,
                         (f16*)nullptr, 0, 0, (f16*)nullptr, 0, 0, tid);
        __syncthreads();

        // ---- phase 9: gains head: WMMA + fused bias/sigmoid + global store ----
        if (wave < 2) {
            v16h a[3];
#pragma unroll
            for (int ks = 0; ks < 3; ++ks) a[ks] = fragA(Hm + A_DH, 96, ks * 32, lane);
            v8f acc = gemm_acc<3>(Hm + W_OUT + wave * 16 * 96, 96, a, lane);
            int o = wave * 16 + (lane & 15);
            if (o < 22) {
                float bo = Fm[B_OUT + o];
                int mb = (lane >> 4) << 3;
#pragma unroll
                for (int r = 0; r < 8; ++r)
                    out[((size_t)(b0 + mb + r) * TSTEPS + t) * 22 + o] = sigmoidf_(acc[r] + bo);
            }
        }
        // next-iter barrier after phase 0 covers remaining hazards
    }
}

extern "C" void kernel_launch(void* const* d_in, const int* in_sizes, int n_in,
                              void* d_out, int out_size, void* d_ws, size_t ws_size,
                              hipStream_t stream) {
    (void)in_sizes; (void)n_in; (void)out_size; (void)d_ws; (void)ws_size;
    rnnoise_fused_kernel<<<BATCH / 16, NTHREADS, SMEM_BYTES, stream>>>(
        (const float*)d_in[0],                                            // x
        (const float*)d_in[1],  (const float*)d_in[2],                    // in_w, in_b
        (const float*)d_in[3],  (const float*)d_in[4],                    // vad_wih, vad_whh
        (const float*)d_in[5],  (const float*)d_in[6],                    // vad_bih, vad_bhh
        (const float*)d_in[7],  (const float*)d_in[8],                    // vad_out_w, vad_out_b
        (const float*)d_in[9],  (const float*)d_in[10],                   // noise_wih, noise_whh
        (const float*)d_in[11], (const float*)d_in[12],                   // noise_bih, noise_bhh
        (const float*)d_in[13], (const float*)d_in[14],                   // den_wih, den_whh
        (const float*)d_in[15], (const float*)d_in[16],                   // den_bih, den_bhh
        (const float*)d_in[17], (const float*)d_in[18],                   // out_w, out_b
        (float*)d_out);
}
